// EnhancedLSTM_65558380806178
// MI455X (gfx1250) — compile-verified, matrix operand order checked by reading, not run on hardware
//
#include <hip/hip_runtime.h>
#include <math.h>

// EnhancedLSTM forward for MI455X (gfx1250, wave32, WMMA).
// bf16 WMMA (16x16x32) + fp32 accumulation for all GEMMs; weights repacked
// once into WMMA-fragment-major layout so the inner loops issue b128 loads;
// CDNA5 async global->LDS double-buffering for the weight stream.

#define B_ 32
#define S_ 512
#define I_ 64
#define H_ 512
#define EPS_ 1e-5f

typedef __attribute__((ext_vector_type(16))) __bf16 v16bf;
typedef __attribute__((ext_vector_type(8)))  float  v8f;
typedef __attribute__((ext_vector_type(8)))  unsigned short us8v;
typedef __attribute__((ext_vector_type(4)))  int    v4i;

#if defined(__has_builtin)
#if __has_builtin(__builtin_amdgcn_global_load_async_to_lds_b128) && \
    __has_builtin(__builtin_amdgcn_s_wait_asynccnt)
#define EL_ASYNC 1
#endif
#endif

#ifdef EL_ASYNC
typedef __attribute__((address_space(1))) v4i as1_v4i;
typedef __attribute__((address_space(3))) v4i as3_v4i;
static __device__ __forceinline__ void el_async_b128(const void* g, void* l) {
  // LDS offset is the low 32 bits of the flat address (ISA 10.2 aperture rule).
  __builtin_amdgcn_global_load_async_to_lds_b128(
      (as1_v4i*)(size_t)g, (as3_v4i*)(unsigned int)(size_t)l, 0, 0);
}
#endif

static __device__ __forceinline__ unsigned short bf_bits(float f) {
  union { float f; unsigned u; } a; a.f = f;
  unsigned r = (a.u + 0x7FFFu + ((a.u >> 16) & 1u)) >> 16;  // RNE
  return (unsigned short)r;
}
static __device__ __forceinline__ __bf16 bits_bf(unsigned short s) {
  union { unsigned short s; __bf16 b; } o; o.s = s; return o.b;
}
static __device__ __forceinline__ __bf16 f2bf(float f) { return bits_bf(bf_bits(f)); }
static __device__ __forceinline__ float sigmoidf_(float x) { return 1.0f / (1.0f + __expf(-x)); }

// Build bf16 A-fragment from 4 float4 runs (k = khalf + {0..7} U {16..23}).
static __device__ __forceinline__ v16bf mk_afrag(float4 x0, float4 x1, float4 x2, float4 x3) {
  v16bf a;
  a[0]=f2bf(x0.x); a[1]=f2bf(x0.y); a[2]=f2bf(x0.z); a[3]=f2bf(x0.w);
  a[4]=f2bf(x1.x); a[5]=f2bf(x1.y); a[6]=f2bf(x1.z); a[7]=f2bf(x1.w);
  a[8]=f2bf(x2.x); a[9]=f2bf(x2.y); a[10]=f2bf(x2.z); a[11]=f2bf(x2.w);
  a[12]=f2bf(x3.x); a[13]=f2bf(x3.y); a[14]=f2bf(x3.z); a[15]=f2bf(x3.w);
  return a;
}

static __device__ __forceinline__ float blk_sum(float v, float* sm) {
  int t = threadIdx.x;
  sm[t] = v; __syncthreads();
  for (int s = 128; s > 0; s >>= 1) { if (t < s) sm[t] += sm[t + s]; __syncthreads(); }
  float r = sm[0]; __syncthreads(); return r;
}
static __device__ __forceinline__ float blk_max(float v, float* sm) {
  int t = threadIdx.x;
  sm[t] = v; __syncthreads();
  for (int s = 128; s > 0; s >>= 1) { if (t < s) sm[t] = fmaxf(sm[t], sm[t + s]); __syncthreads(); }
  float r = sm[0]; __syncthreads(); return r;
}

// ---------------------------------------------------------------- utilities
__global__ void el_zero(float* __restrict__ p, size_t n) {
  size_t i = (size_t)blockIdx.x * 256 + threadIdx.x;
  if (i < n) p[i] = 0.0f;
}

// Repack row-major fp32 weight [K,N] into WMMA-fragment-major bf16:
// index = (((ntile*KC + kc)*32 + lane)*16 + e), lane's 16 values contiguous.
__global__ void el_pack_b(const float* __restrict__ W, unsigned short* __restrict__ out,
                          int N, int K) {
  size_t idx = (size_t)blockIdx.x * 256 + threadIdx.x;
  if (idx >= (size_t)K * N) return;
  int e    = (int)(idx & 15);
  int lane = (int)((idx >> 4) & 31);
  size_t rest = idx >> 9;
  int KC = K >> 5;
  int kc = (int)(rest % KC);
  int nt = (int)(rest / KC);
  int khalf = (lane >> 4) << 3;
  int k = kc * 32 + khalf + (e < 8 ? e : 8 + e);
  int n = nt * 16 + (lane & 15);
  out[idx] = bf_bits(W[(size_t)k * N + n]);
}

// conv weight [O=512,I=512,T=3] -> fragment-packed bf16 with k = tap*512+i.
__global__ void el_pack_conv(const float* __restrict__ W, unsigned short* __restrict__ out) {
  size_t idx = (size_t)blockIdx.x * 256 + threadIdx.x;  // 786432
  int e    = (int)(idx & 15);
  int lane = (int)((idx >> 4) & 31);
  size_t rest = idx >> 9;
  const int KC = 48;                                     // 1536/32
  int kc = (int)(rest % KC);
  int nt = (int)(rest / KC);
  int khalf = (lane >> 4) << 3;
  int k = kc * 32 + khalf + (e < 8 ? e : 8 + e);
  int n = nt * 16 + (lane & 15);                         // output channel
  out[idx] = bf_bits(W[((size_t)n * 512 + (k & 511)) * 3 + (k >> 9)]);
}

// V [B,S,H] bf16 -> Vt [B,NH,64,S] bf16 (contiguous over sequence)
__global__ void el_vtrans(const unsigned short* __restrict__ vin,
                          unsigned short* __restrict__ vout) {
  size_t idx = (size_t)blockIdx.x * 256 + threadIdx.x;   // 8388608
  int s  = (int)(idx & 511);
  int d  = (int)((idx >> 9) & 63);
  int hb = (int)(idx >> 15);                             // b*8+hh
  int hh = hb & 7, b = hb >> 3;
  vout[idx] = vin[((size_t)(b * 512 + s)) * 512 + hh * 64 + d];
}

__global__ void el_add_pe(float* __restrict__ h) {
  size_t idx = (size_t)blockIdx.x * 256 + threadIdx.x;   // B*S*H
  int hh = (int)(idx & 511);
  int s  = (int)((idx >> 9) & 511);
  int h2 = hh & ~1;
  float div = __expf(-(float)h2 * (9.2103403719761836f / 512.0f)); // ln(1e4)/H
  float ang = (float)s * div;
  h[idx] += (hh & 1) ? __cosf(ang) : __sinf(ang);
}

// --------------------------------------------------------- generic WMMA GEMM
// C[M,N] = act(A[M,K](f32) @ Bp(packed bf16) + bias) (+resid). 16x16 tile/wave.
__global__ void el_gemm_wmma(const float* __restrict__ A, int lda,
                             const unsigned short* __restrict__ Bp,
                             const float* __restrict__ bias,
                             float* __restrict__ Cf, unsigned short* __restrict__ Cb, int ldc,
                             const float* __restrict__ resid, int ldr,
                             int M, int N, int K, int act) {
#ifdef EL_ASYNC
  __shared__ __align__(16) unsigned short bstage[8][2][512];
#endif
  int wid  = threadIdx.x >> 5;
  int lane = threadIdx.x & 31;
  int wave = blockIdx.x * 8 + wid;
  int ntn  = N >> 4;
  int KC   = K >> 5;
  int total = (M >> 4) * ntn;
  if (wave >= total) return;                 // wave-uniform exit, EXEC stays full
  int tmi = wave / ntn, tni = wave % ntn;
  int tm = tmi << 4;
  int n  = (tni << 4) + (lane & 15);
  int m  = tm + (lane & 15);
  int khalf = (lane >> 4) << 3;
  const float* Arow = A + (size_t)m * lda;
  const unsigned short* pb = Bp + (size_t)tni * KC * 512 + lane * 16;
  v8f c = {};
#ifdef EL_ASYNC
  el_async_b128(pb, &bstage[wid][0][lane * 16]);
  el_async_b128(pb + 8, &bstage[wid][0][lane * 16 + 8]);
#endif
  for (int kc = 0; kc < KC; ++kc) {
    int k0 = kc << 5;
    const float4* ap = (const float4*)(Arow + k0 + khalf);
    float4 x0 = ap[0], x1 = ap[1], x2 = ap[4], x3 = ap[5];
    v16bf a = mk_afrag(x0, x1, x2, x3);
    union { us8v h[2]; v16bf v; } ub;
#ifdef EL_ASYNC
    if (kc + 1 < KC) {
      const unsigned short* pn = pb + (size_t)(kc + 1) * 512;
      unsigned short* ls = &bstage[wid][(kc + 1) & 1][lane * 16];
      el_async_b128(pn, ls);
      el_async_b128(pn + 8, ls + 8);
      __builtin_amdgcn_s_wait_asynccnt(2);
    } else {
      __builtin_amdgcn_s_wait_asynccnt(0);
    }
    const us8v* bsrc = (const us8v*)&bstage[wid][kc & 1][lane * 16];
    ub.h[0] = bsrc[0]; ub.h[1] = bsrc[1];
#else
    const us8v* bsrc = (const us8v*)(pb + (size_t)kc * 512);
    __builtin_prefetch(pb + (size_t)(kc + 1) * 512, 0, 1);  // global_prefetch_b8
    ub.h[0] = bsrc[0]; ub.h[1] = bsrc[1];
#endif
    c = __builtin_amdgcn_wmma_f32_16x16x32_bf16(false, a, false, ub.v, (short)0, c, false, false);
  }
  float bn = bias ? bias[n] : 0.0f;
#pragma unroll
  for (int r = 0; r < 8; ++r) {
    int mr = tm + r + ((lane >> 4) << 3);
    float v = c[r] + bn;
    if (act == 1) v = fmaxf(v, 0.0f);
    else if (act == 2) v = tanhf(v);
    if (resid) v += resid[(size_t)mr * ldr + n];
    if (Cf) Cf[(size_t)mr * ldc + n] = v;
    if (Cb) Cb[(size_t)mr * ldc + n] = bf_bits(v);
  }
}

// ------------------------------------------------------------- LSTM kernels
// gates[32,2048] = concat(xin[32,512], hprev[32,512]) @ Wg(packed) + b
__global__ void el_lstm_gates(const float* __restrict__ xin, int ldx,
                              const float* __restrict__ hprev,
                              const unsigned short* __restrict__ Bp,
                              const float* __restrict__ bias,
                              float* __restrict__ out) {
#ifdef EL_ASYNC
  __shared__ __align__(16) unsigned short bstage[8][2][512];
#endif
  const int N = 2048, KC = 32;               // K = 1024
  int wid  = threadIdx.x >> 5;
  int lane = threadIdx.x & 31;
  int wave = blockIdx.x * 8 + wid;           // 0..255
  int tm = (wave >> 7) << 4;                 // 2 m-tiles
  int tni = wave & 127;                      // 128 n-tiles
  int n = (tni << 4) + (lane & 15);
  int m = tm + (lane & 15);
  int khalf = (lane >> 4) << 3;
  const unsigned short* pb = Bp + (size_t)tni * KC * 512 + lane * 16;
  v8f c = {};
#ifdef EL_ASYNC
  el_async_b128(pb, &bstage[wid][0][lane * 16]);
  el_async_b128(pb + 8, &bstage[wid][0][lane * 16 + 8]);
#endif
  for (int kc = 0; kc < KC; ++kc) {
    int k0 = kc << 5;
    const float* base = (k0 < 512) ? (xin + (size_t)m * ldx + k0)
                                   : (hprev + (size_t)m * 512 + (k0 - 512));
    const float4* ap = (const float4*)(base + khalf);
    float4 x0 = ap[0], x1 = ap[1], x2 = ap[4], x3 = ap[5];
    v16bf a = mk_afrag(x0, x1, x2, x3);
    union { us8v h[2]; v16bf v; } ub;
#ifdef EL_ASYNC
    if (kc + 1 < KC) {
      const unsigned short* pn = pb + (size_t)(kc + 1) * 512;
      unsigned short* ls = &bstage[wid][(kc + 1) & 1][lane * 16];
      el_async_b128(pn, ls);
      el_async_b128(pn + 8, ls + 8);
      __builtin_amdgcn_s_wait_asynccnt(2);
    } else {
      __builtin_amdgcn_s_wait_asynccnt(0);
    }
    const us8v* bsrc = (const us8v*)&bstage[wid][kc & 1][lane * 16];
    ub.h[0] = bsrc[0]; ub.h[1] = bsrc[1];
#else
    const us8v* bsrc = (const us8v*)(pb + (size_t)kc * 512);
    __builtin_prefetch(pb + (size_t)(kc + 1) * 512, 0, 1);
    ub.h[0] = bsrc[0]; ub.h[1] = bsrc[1];
#endif
    c = __builtin_amdgcn_wmma_f32_16x16x32_bf16(false, a, false, ub.v, (short)0, c, false, false);
  }
  float bn = bias[n];
#pragma unroll
  for (int r = 0; r < 8; ++r) {
    int mr = tm + r + ((lane >> 4) << 3);
    out[(size_t)mr * N + n] = c[r] + bn;
  }
}

// cell + LN1 + LN2 (per batch row). grid=B, block=256, 2 channels/thread.
__global__ void el_lstm_cell(const float* __restrict__ gates, float* __restrict__ cst,
                             const float* __restrict__ g1, const float* __restrict__ b1,
                             const float* __restrict__ g2, const float* __restrict__ b2,
                             float* __restrict__ hln) {
  __shared__ float sm[256];
  int b = blockIdx.x, t = threadIdx.x;
  const float* gr = gates + (size_t)b * 2048;
  int ch0 = t, ch1 = t + 256;
  float og0, og1, cn0, cn1;
  {
    float ig = sigmoidf_(gr[ch0]);
    float fg = sigmoidf_(gr[512 + ch0]);
    float gg = tanhf(gr[1024 + ch0]);
    og0 = sigmoidf_(gr[1536 + ch0]);
    cn0 = fg * cst[b * 512 + ch0] + ig * gg;
  }
  {
    float ig = sigmoidf_(gr[ch1]);
    float fg = sigmoidf_(gr[512 + ch1]);
    float gg = tanhf(gr[1024 + ch1]);
    og1 = sigmoidf_(gr[1536 + ch1]);
    cn1 = fg * cst[b * 512 + ch1] + ig * gg;
  }
  float s  = blk_sum(cn0 + cn1, sm);
  float mean = s * (1.0f / 512.0f);
  float sq = blk_sum(cn0 * cn0 + cn1 * cn1, sm);
  float var = sq * (1.0f / 512.0f) - mean * mean;
  float rs = rsqrtf(var + EPS_);
  float cl0 = (cn0 - mean) * rs * g1[ch0] + b1[ch0];
  float cl1 = (cn1 - mean) * rs * g1[ch1] + b1[ch1];
  cst[b * 512 + ch0] = cl0;
  cst[b * 512 + ch1] = cl1;
  float h0 = og0 * tanhf(cl0);
  float h1 = og1 * tanhf(cl1);
  s = blk_sum(h0 + h1, sm);   mean = s * (1.0f / 512.0f);
  sq = blk_sum(h0 * h0 + h1 * h1, sm);
  var = sq * (1.0f / 512.0f) - mean * mean;
  rs = rsqrtf(var + EPS_);
  hln[b * 512 + ch0] = (h0 - mean) * rs * g2[ch0] + b2[ch0];
  hln[b * 512 + ch1] = (h1 - mean) * rs * g2[ch1] + b2[ch1];
}

__global__ void el_lstm_combine(const float* __restrict__ glu, const float* __restrict__ skip,
                                float* __restrict__ hstate, float* __restrict__ dest, int ldd) {
  int b = blockIdx.x, t = threadIdx.x;
#pragma unroll
  for (int u = 0; u < 2; ++u) {
    int c = t + u * 256;
    float a = glu[(size_t)b * 1024 + c];
    float g = glu[(size_t)b * 1024 + 512 + c];
    float v = a * sigmoidf_(g) + skip[b * 512 + c];
    hstate[b * 512 + c] = v;
    dest[(size_t)b * ldd + c] = v;
  }
}

// ----------------------------------------------------- fused self-attention
__global__ void el_attn(const unsigned short* __restrict__ Qb,
                        const unsigned short* __restrict__ Kb,
                        const unsigned short* __restrict__ Vt,
                        float* __restrict__ ctx) {
  __shared__ __align__(16) float sc[16][512];   // 32KB score/prob block
  __shared__ float red[16][17];
  __shared__ float pout[4][16][16];
  int id = blockIdx.x;
  int qt = id & 31;
  int hh = (id >> 5) & 7;
  int b  = id >> 8;
  int lane = threadIdx.x & 31;
  int wave = threadIdx.x >> 5;
  int khalf = (lane >> 4) << 3;
  const unsigned short* qrow =
      Qb + ((size_t)(b * S_ + qt * 16 + (lane & 15))) * H_ + hh * 64;

  // Phase 1: S = (Q K^T)/8
  for (int kt = wave; kt < 32; kt += 8) {
    v8f c = {};
    const unsigned short* krow =
        Kb + ((size_t)(b * S_ + kt * 16 + (lane & 15))) * H_ + hh * 64;
    for (int d0 = 0; d0 < 64; d0 += 32) {
      union { us8v h[2]; v16bf v; } ua, ub;
      ua.h[0] = *(const us8v*)(qrow + d0 + khalf);
      ua.h[1] = *(const us8v*)(qrow + d0 + khalf + 16);
      ub.h[0] = *(const us8v*)(krow + d0 + khalf);
      ub.h[1] = *(const us8v*)(krow + d0 + khalf + 16);
      c = __builtin_amdgcn_wmma_f32_16x16x32_bf16(false, ua.v, false, ub.v, (short)0, c, false, false);
    }
#pragma unroll
    for (int r = 0; r < 8; ++r)
      sc[r + ((lane >> 4) << 3)][kt * 16 + (lane & 15)] = c[r] * 0.125f;
  }
  __syncthreads();

  // Phase 2: softmax over 512 keys per row (16 threads per row)
  {
    int row = threadIdx.x >> 4;
    int sub = threadIdx.x & 15;
    float mx = -3.0e38f;
    for (int cc = sub; cc < 512; cc += 16) mx = fmaxf(mx, sc[row][cc]);
    red[row][sub] = mx; __syncthreads();
    if (sub == 0) { float m2 = red[row][0]; for (int i = 1; i < 16; ++i) m2 = fmaxf(m2, red[row][i]); red[row][16] = m2; }
    __syncthreads();
    float rmax = red[row][16];
    float sum = 0.0f;
    for (int cc = sub; cc < 512; cc += 16) { float e = __expf(sc[row][cc] - rmax); sc[row][cc] = e; sum += e; }
    red[row][sub] = sum; __syncthreads();
    if (sub == 0) { float s2 = 0.0f; for (int i = 0; i < 16; ++i) s2 += red[row][i]; red[row][16] = s2; }
    __syncthreads();
    float inv = 1.0f / red[row][16];
    for (int cc = sub; cc < 512; cc += 16) sc[row][cc] *= inv;
  }
  __syncthreads();

  // Phase 3: out = P @ V, 4 n-tiles x 2-way K split
  {
    int nt = wave & 3;
    int kh = wave >> 2;
    const float* srow = &sc[lane & 15][0];
    const unsigned short* vrow =
        Vt + ((size_t)((b * 8 + hh) * 64 + nt * 16 + (lane & 15))) * 512;
    v8f c = {};
    for (int kc = kh * 8; kc < kh * 8 + 8; ++kc) {
      int k0 = kc * 32;
      const float4* sp = (const float4*)(srow + k0 + khalf);
      v16bf a = mk_afrag(sp[0], sp[1], sp[4], sp[5]);
      union { us8v h[2]; v16bf v; } ub;
      ub.h[0] = *(const us8v*)(vrow + k0 + khalf);
      ub.h[1] = *(const us8v*)(vrow + k0 + khalf + 16);
      c = __builtin_amdgcn_wmma_f32_16x16x32_bf16(false, a, false, ub.v, (short)0, c, false, false);
    }
    if (kh == 1) {
#pragma unroll
      for (int r = 0; r < 8; ++r) pout[nt][r + ((lane >> 4) << 3)][lane & 15] = c[r];
    }
    __syncthreads();
    if (kh == 0) {
#pragma unroll
      for (int r = 0; r < 8; ++r) {
        int mr = r + ((lane >> 4) << 3);
        float v = c[r] + pout[nt][mr][lane & 15];
        ctx[((size_t)(b * S_ + qt * 16 + mr)) * H_ + hh * 64 + nt * 16 + (lane & 15)] = v;
      }
    }
  }
}

// -------------------------------------------- temporal conv as K=1536 WMMA
__global__ void el_conv(const float* __restrict__ src, const unsigned short* __restrict__ Wp,
                        const float* __restrict__ bias, float* __restrict__ dst, int relu) {
  __shared__ __align__(16) unsigned short patch[18][512];   // s0-1..s0+16, bf16
  int id = blockIdx.x;
  int ot = id & 7;
  int st = (id >> 3) & 31;
  int b  = id >> 8;
  int tid = threadIdx.x;                       // 128 threads, 4 waves
  for (int idx = tid; idx < 18 * 512; idx += 128) {
    int p = idx >> 9, chn = idx & 511;
    int s = st * 16 + p - 1;
    float v = (s >= 0 && s < S_) ? src[((size_t)(b * S_ + s)) * H_ + chn] : 0.0f;
    patch[p][chn] = bf_bits(v);
  }
  __syncthreads();
  int lane = tid & 31;
  int wv = tid >> 5;
  int khalf = (lane >> 4) << 3;
  int n = ot * 64 + wv * 16 + (lane & 15);
  const unsigned short* pb = Wp + (size_t)(ot * 4 + wv) * 48 * 512 + lane * 16;
  v8f c = {};
  for (int kc = 0; kc < 48; ++kc) {
    int k0 = kc << 5;
    int tap = k0 >> 9, ch0 = (k0 & 511) + khalf;
    const unsigned short* prow = &patch[(lane & 15) + tap][ch0];
    union { us8v h[2]; v16bf v; } ua, ub;
    ua.h[0] = *(const us8v*)(prow);
    ua.h[1] = *(const us8v*)(prow + 16);
    const us8v* bsrc = (const us8v*)(pb + (size_t)kc * 512);
    __builtin_prefetch(pb + (size_t)(kc + 1) * 512, 0, 1);
    ub.h[0] = bsrc[0]; ub.h[1] = bsrc[1];
    c = __builtin_amdgcn_wmma_f32_16x16x32_bf16(false, ua.v, false, ub.v, (short)0, c, false, false);
  }
  float bn = bias[n];
#pragma unroll
  for (int r = 0; r < 8; ++r) {
    int mr = r + ((lane >> 4) << 3);
    float v = c[r] + bn;
    if (relu) v = fmaxf(v, 0.0f);
    dst[((size_t)(b * S_ + st * 16 + mr)) * H_ + n] = v;
  }
}

// tcn LN, +residual, final LN; writes ws hidden and d_out hidden
__global__ void el_tcn_final_ln(const float* __restrict__ y2, float* __restrict__ hid,
                                float* __restrict__ outh,
                                const float* __restrict__ tg, const float* __restrict__ tb,
                                const float* __restrict__ fg, const float* __restrict__ fb) {
  __shared__ float sm[256];
  size_t row = blockIdx.x;
  int t = threadIdx.x;
  const float* yr = y2 + row * 512;
  float v0 = yr[t], v1 = yr[t + 256];
  float s  = blk_sum(v0 + v1, sm);
  float mean = s * (1.0f / 512.0f);
  float sq = blk_sum(v0 * v0 + v1 * v1, sm);
  float var = sq * (1.0f / 512.0f) - mean * mean;
  float rs = rsqrtf(var + EPS_);
  float n0 = (v0 - mean) * rs * tg[t] + tb[t];
  float n1 = (v1 - mean) * rs * tg[t + 256] + tb[t + 256];
  float* hr = hid + row * 512;
  n0 += hr[t]; n1 += hr[t + 256];
  s = blk_sum(n0 + n1, sm);  mean = s * (1.0f / 512.0f);
  sq = blk_sum(n0 * n0 + n1 * n1, sm);
  var = sq * (1.0f / 512.0f) - mean * mean;
  rs = rsqrtf(var + EPS_);
  float o0 = (n0 - mean) * rs * fg[t] + fb[t];
  float o1 = (n1 - mean) * rs * fg[t + 256] + fb[t + 256];
  hr[t] = o0; hr[t + 256] = o1;
  float* orow = outh + row * 512;
  orow[t] = o0; orow[t + 256] = o1;
}

// ------------------------------------------------------------ pooling attn
__global__ void el_pool_logits(const float* __restrict__ t1, const float* __restrict__ w2,
                               const float* __restrict__ b2, float* __restrict__ logits) {
  int row = blockIdx.x * 8 + (threadIdx.x >> 5);
  int lane = threadIdx.x & 31;
  const float* r = t1 + (size_t)row * 512;
  float s = 0.0f;
  for (int i = lane; i < 512; i += 32) s += r[i] * w2[i];
  for (int m = 16; m > 0; m >>= 1) s += __shfl_xor(s, m, 32);
  if (lane == 0) logits[row] = s + b2[0];
}
__global__ void el_pool_softmax(const float* __restrict__ logits, float* __restrict__ aw) {
  __shared__ float sm[256];
  int b = blockIdx.x, t = threadIdx.x;
  float l0 = logits[b * 512 + t], l1 = logits[b * 512 + t + 256];
  float mx = blk_max(fmaxf(l0, l1), sm);
  float e0 = __expf(l0 - mx), e1 = __expf(l1 - mx);
  float sum = blk_sum(e0 + e1, sm);
  float inv = 1.0f / sum;
  aw[b * 512 + t] = e0 * inv;
  aw[b * 512 + t + 256] = e1 * inv;
}
__global__ void el_pool_attend(const float* __restrict__ aw, const float* __restrict__ hid,
                               float* __restrict__ out) {
  int b = blockIdx.x, t = threadIdx.x;
  float a0 = 0.0f, a1 = 0.0f;
  for (int s = 0; s < 512; ++s) {
    float a = aw[b * 512 + s];
    const float* row = hid + ((size_t)(b * 512 + s)) * 512;
    a0 += a * row[t];
    a1 += a * row[t + 256];
  }
  out[b * 512 + t] = a0;
  out[b * 512 + t + 256] = a1;
}

// ================================================================== launch
extern "C" void kernel_launch(void* const* d_in, const int* in_sizes, int n_in,
                              void* d_out, int out_size, void* d_ws, size_t ws_size,
                              hipStream_t stream) {
  (void)in_sizes; (void)n_in; (void)out_size; (void)ws_size;
  const float* x       = (const float*)d_in[0];
  const float* emb_w   = (const float*)d_in[1];
  const float* emb_b   = (const float*)d_in[2];
  const float* gates_w = (const float*)d_in[3];
  const float* gates_b = (const float*)d_in[4];
  const float* ln1_g   = (const float*)d_in[5];
  const float* ln1_b   = (const float*)d_in[6];
  const float* ln2_g   = (const float*)d_in[7];
  const float* ln2_b   = (const float*)d_in[8];
  const float* glu_w   = (const float*)d_in[9];
  const float* glu_b   = (const float*)d_in[10];
  const float* skip_w  = (const float*)d_in[11];
  const float* skip_b  = (const float*)d_in[12];
  const float* q_w = (const float*)d_in[13]; const float* q_b = (const float*)d_in[14];
  const float* k_w = (const float*)d_in[15]; const float* k_b = (const float*)d_in[16];
  const float* v_w = (const float*)d_in[17]; const float* v_b = (const float*)d_in[18];
  const float* o_w = (const float*)d_in[19]; const float* o_b = (const float*)d_in[20];
  const float* conv1_w = (const float*)d_in[21]; const float* conv1_b = (const float*)d_in[22];
  const float* conv2_w = (const float*)d_in[23]; const float* conv2_b = (const float*)d_in[24];
  const float* tcn_g = (const float*)d_in[25]; const float* tcn_b = (const float*)d_in[26];
  const float* fin_g = (const float*)d_in[27]; const float* fin_b = (const float*)d_in[28];
  const float* att1_w = (const float*)d_in[29]; const float* att1_b = (const float*)d_in[30];
  const float* att2_w = (const float*)d_in[31]; const float* att2_b = (const float*)d_in[32];

  float* out_att = (float*)d_out;                     // [B,H]
  float* out_hid = out_att + B_ * H_;                 // [B,S,H]
  float* out_aw  = out_hid + (size_t)B_ * S_ * H_;    // [B,S,1]

  size_t off = 0;
  auto alloc_f = [&](size_t n) -> float* {
    off = (off + 255) & ~(size_t)255;
    float* p = (float*)((char*)d_ws + off); off += n * sizeof(float); return p;
  };
  auto alloc_h = [&](size_t n) -> unsigned short* {
    off = (off + 255) & ~(size_t)255;
    unsigned short* p = (unsigned short*)((char*)d_ws + off); off += n * sizeof(unsigned short); return p;
  };
  const size_t BSH = (size_t)B_ * S_ * H_;            // 8388608
  float* hidden  = alloc_f(BSH);
  float* ctx     = alloc_f(BSH);
  float* y1      = alloc_f(BSH);
  float* y2      = alloc_f(BSH);
  float* logits  = alloc_f(B_ * S_);
  float* gatesbf = alloc_f(B_ * 2048);
  float* hstate  = alloc_f(2 * B_ * H_);
  float* cstate  = alloc_f(2 * B_ * H_);
  float* hln     = alloc_f(B_ * H_);
  float* gluout  = alloc_f(B_ * 1024);
  float* skipout = alloc_f(B_ * H_);
  float* l0out   = alloc_f(B_ * H_);
  unsigned short* Qb = alloc_h(BSH);
  unsigned short* Kb = alloc_h(BSH);
  unsigned short* Vb = alloc_h(BSH);
  unsigned short* Vt = alloc_h(BSH);
  unsigned short* wb_emb   = alloc_h(64 * 512);
  unsigned short* wb_gates = alloc_h((size_t)2 * 1024 * 2048);
  unsigned short* wb_glu   = alloc_h(512 * 1024);
  unsigned short* wb_skip  = alloc_h(2 * 512 * 512);
  unsigned short* wb_q = alloc_h(512 * 512);
  unsigned short* wb_k = alloc_h(512 * 512);
  unsigned short* wb_v = alloc_h(512 * 512);
  unsigned short* wb_o = alloc_h(512 * 512);
  unsigned short* wb_a1 = alloc_h(512 * 512);
  unsigned short* wb_c1 = alloc_h(3 * 512 * 512);
  unsigned short* wb_c2 = alloc_h(3 * 512 * 512);

  auto pack = [&](const float* W, unsigned short* dst, int N, int K) {
    size_t n = (size_t)N * K;
    el_pack_b<<<(unsigned)((n + 255) / 256), 256, 0, stream>>>(W, dst, N, K);
  };
  auto gemm = [&](const float* A, int lda, const unsigned short* Bp, const float* bias,
                  float* Cf, unsigned short* Cb, int ldc, const float* resid, int ldr,
                  int M, int N, int K, int act) {
    int waves = (M / 16) * (N / 16);
    int blocks = (waves + 7) / 8;
    el_gemm_wmma<<<blocks, 256, 0, stream>>>(A, lda, Bp, bias, Cf, Cb, ldc, resid, ldr, M, N, K, act);
  };

  // --- weight repack (bf16 fragment-major; L2-resident across the scan) ---
  pack(emb_w, wb_emb, 512, 64);
  pack(gates_w, wb_gates, 2048, 1024);
  pack(gates_w + (size_t)1024 * 2048, wb_gates + (size_t)1024 * 2048, 2048, 1024);
  pack(glu_w, wb_glu, 1024, 512);
  pack(skip_w, wb_skip, 512, 512);
  pack(skip_w + 512 * 512, wb_skip + 512 * 512, 512, 512);
  pack(q_w, wb_q, 512, 512);  pack(k_w, wb_k, 512, 512);
  pack(v_w, wb_v, 512, 512);  pack(o_w, wb_o, 512, 512);
  pack(att1_w, wb_a1, 512, 512);
  el_pack_conv<<<3072, 256, 0, stream>>>(conv1_w, wb_c1);
  el_pack_conv<<<3072, 256, 0, stream>>>(conv2_w, wb_c2);
  el_zero<<<128, 256, 0, stream>>>(hstate, 2 * B_ * H_);
  el_zero<<<128, 256, 0, stream>>>(cstate, 2 * B_ * H_);

  // --- embedding + positional encoding ---
  gemm(x, I_, wb_emb, emb_b, hidden, nullptr, H_, nullptr, 0, B_ * S_, H_, I_, 0);
  el_add_pe<<<32768, 256, 0, stream>>>(hidden);

  // --- 2-layer LSTM scan (stream-ordered recurrence) ---
  for (int t = 0; t < S_; ++t) {
    const float* xin0 = hidden + (size_t)t * H_;        // row stride S*H
    // layer 0
    el_lstm_gates<<<32, 256, 0, stream>>>(xin0, S_ * H_, hstate, wb_gates, gates_b, gatesbf);
    el_lstm_cell<<<32, 256, 0, stream>>>(gatesbf, cstate, ln1_g, ln1_b, ln2_g, ln2_b, hln);
    gemm(hln, H_, wb_glu, glu_b, gluout, nullptr, 1024, nullptr, 0, B_, 1024, H_, 0);
    gemm(xin0, S_ * H_, wb_skip, skip_b, skipout, nullptr, H_, nullptr, 0, B_, H_, H_, 0);
    el_lstm_combine<<<32, 256, 0, stream>>>(gluout, skipout, hstate, l0out, H_);
    // layer 1
    el_lstm_gates<<<32, 256, 0, stream>>>(l0out, H_, hstate + B_ * H_,
                                          wb_gates + (size_t)1024 * 2048, gates_b + 2048, gatesbf);
    el_lstm_cell<<<32, 256, 0, stream>>>(gatesbf, cstate + B_ * H_,
                                         ln1_g + 512, ln1_b + 512, ln2_g + 512, ln2_b + 512, hln);
    gemm(hln, H_, wb_glu, glu_b, gluout, nullptr, 1024, nullptr, 0, B_, 1024, H_, 0);
    gemm(l0out, H_, wb_skip + 512 * 512, skip_b + 512, skipout, nullptr, H_, nullptr, 0, B_, H_, H_, 0);
    el_lstm_combine<<<32, 256, 0, stream>>>(gluout, skipout, hstate + B_ * H_,
                                            hidden + (size_t)t * H_, S_ * H_);
  }

  // --- multi-head self-attention + residual ---
  gemm(hidden, H_, wb_q, q_b, nullptr, Qb, H_, nullptr, 0, B_ * S_, H_, H_, 0);
  gemm(hidden, H_, wb_k, k_b, nullptr, Kb, H_, nullptr, 0, B_ * S_, H_, H_, 0);
  gemm(hidden, H_, wb_v, v_b, nullptr, Vb, H_, nullptr, 0, B_ * S_, H_, H_, 0);
  el_vtrans<<<32768, 256, 0, stream>>>(Vb, Vt);
  el_attn<<<8192, 256, 0, stream>>>(Qb, Kb, Vt, ctx);
  gemm(ctx, H_, wb_o, o_b, hidden, nullptr, H_, hidden, H_, B_ * S_, H_, H_, 0);

  // --- temporal conv block + double LayerNorm ---
  el_conv<<<8192, 128, 0, stream>>>(hidden, wb_c1, conv1_b, y1, 1);
  el_conv<<<8192, 128, 0, stream>>>(y1, wb_c2, conv2_b, y2, 0);
  el_tcn_final_ln<<<16384, 256, 0, stream>>>(y2, hidden, out_hid, tcn_g, tcn_b, fin_g, fin_b);

  // --- attention pooling over sequence ---
  gemm(hidden, H_, wb_a1, att1_b, y1, nullptr, H_, nullptr, 0, B_ * S_, H_, H_, 2); // tanh
  el_pool_logits<<<2048, 256, 0, stream>>>(y1, att2_w, att2_b, logits);
  el_pool_softmax<<<32, 256, 0, stream>>>(logits, out_aw);
  el_pool_attend<<<32, 256, 0, stream>>>(out_aw, hidden, out_att);
}
